// ToksIntraDistillation_62405874811833
// MI455X (gfx1250) — matrix operand order repaired and checked by reading, not compiled, not run on hardware
//
#include <hip/hip_runtime.h>

#define HIDDEN 256
#define SEQ 233
#define NSEG 3

typedef unsigned short u16;
typedef unsigned int   u32;

typedef __attribute__((ext_vector_type(16))) __bf16 v16bf;
typedef __attribute__((ext_vector_type(8)))  float  v8f;
typedef __attribute__((ext_vector_type(4)))  u32    u32x4;
typedef __attribute__((ext_vector_type(4)))  u16    u16x4;
typedef __attribute__((ext_vector_type(4)))  float  f32x4;

// f32 -> bf16 via native convert (lowers to v_cvt_pk_bf16_f32 on gfx1250)
__device__ __forceinline__ u16 f2bf(float f) {
  return __builtin_bit_cast(u16, static_cast<__bf16>(f));
}

union FragAB {
  v16bf v;
  u32x4 q[2];
};

// A fragment (16x32 bf16, rows=output rows), row-major source [row][k]:
//   lane L(0..15): row M=L, K = kBase+[0..7] (q0) and kBase+[16..23] (q1)
//   lane L+16:     row M=L, K = kBase+[8..15]     and kBase+[24..31]
__device__ __forceinline__ v16bf load_a(const u16* buf, int pitch, int rowBase,
                                        int kBase, int lane) {
  const u16* p = buf + (rowBase + (lane & 15)) * pitch + kBase + ((lane >> 4) << 3);
  FragAB f;
  f.q[0] = *(const u32x4*)(p);
  f.q[1] = *(const u32x4*)(p + 16);
  return f.v;
}

// B fragment (32x16 bf16, cols=output cols), source stored row-major [col][k]:
//   lane L(0..15): col N=L, K = kBase+[0..7],[16..23]; lane L+16: col N=L, K = kBase+[8..15],[24..31]
__device__ __forceinline__ v16bf load_b(const u16* buf, int pitch, int colBase,
                                        int kBase, int lane) {
  const u16* p = buf + (colBase + (lane & 15)) * pitch + kBase + ((lane >> 4) << 3);
  FragAB f;
  f.q[0] = *(const u32x4*)(p);
  f.q[1] = *(const u32x4*)(p + 16);
  return f.v;
}

__device__ __forceinline__ v8f wmma_bf16(v16bf a, v16bf b, v8f c) {
  return __builtin_amdgcn_wmma_f32_16x16x32_bf16(false, a, false, b, (short)0, c,
                                                 false, false);
}

// ---- prologue: f32 weights -> bf16 in workspace (L2-resident, broadcast) ----
__global__ void wcvt_kernel(const float* __restrict__ W0, const float* __restrict__ W1,
                            const float* __restrict__ W2, const float* __restrict__ W3,
                            u16* __restrict__ dst) {
  const int idx = blockIdx.x * 256 + threadIdx.x;   // 4*65536 total
  const float* Ws[4] = {W0, W1, W2, W3};
  dst[idx] = f2bf(Ws[idx >> 16][idx & 65535]);
}

// LDS layout (u16 elements):
//   xs   [128][264]  @0       (67,584 B)  x in bf16, lives whole kernel
//   r1   [128][264]  @33792   (67,584 B)  s1; reused as -probs [128][136]
//   r2   [256][136]  @67584   (69,632 B)  s2 [128][264]; reused as s3^T [256][136]
#define PX 264
#define PT 136
#define PP 136

__global__ __launch_bounds__(256, 1) void fused_attn_kernel(
    const float* __restrict__ toks, const u16* __restrict__ Wbf,
    const float* __restrict__ b0, const float* __restrict__ b1,
    const float* __restrict__ b2, const float* __restrict__ b3,
    float* __restrict__ out) {
  __shared__ __align__(16) u16 smem[102400];   // 204,800 B
  u16* xs = smem;
  u16* s1 = smem + 33792;
  u16* s2 = smem + 67584;
  u16* probs = s1;   // overlays s1 after scores consumed
  u16* s3T   = s2;   // overlays s2 after scores consumed

  const int blk = blockIdx.x;
  const int b   = blk / NSEG;
  const int seg = blk - b * NSEG;
  const int segStart = (seg == 0) ? 0 : (seg == 1 ? 103 : 222);
  const int N        = (seg == 0) ? 103 : (seg == 1 ? 119 : 11);
  const int nT   = (N + 15) >> 4;
  const int Npad = nT << 4;

  const int tid  = threadIdx.x;
  const int lane = tid & 31;
  const int wave = tid >> 5;
  const int half = lane >> 4;
  const int ln   = lane & 15;
  const int rT   = wave;       // wave <-> row-tile (scores/softmax phases)
  const int oT0  = wave * 2;   // wave <-> column-tile pair (weight GEMM phases)

  // ---- Phase 0: toks f32 -> xs bf16 (zero-pad rows >= N) ----
  const float* xg = toks + (b * SEQ + segStart) * HIDDEN;
  for (int idx = tid; idx < Npad * (HIDDEN / 4); idx += 256) {
    const int row = idx >> 6;
    const int c4  = (idx & 63) << 2;
    f32x4 v = {0.f, 0.f, 0.f, 0.f};
    if (row < N) v = *(const f32x4*)(xg + row * HIDDEN + c4);
    u16x4 h;
    h.x = f2bf(v.x); h.y = f2bf(v.y); h.z = f2bf(v.z); h.w = f2bf(v.w);
    *(u16x4*)(xs + row * PX + c4) = h;
  }
  __syncthreads();

  // ---- Phase 1: s1 = x@W1^T + b1, s2 = x@W2^T + b2 (bf16, row-major LDS) ----
  // Wave owns 2 column tiles: weight B-fragments cached in regs -> each weight
  // matrix is read from L2 exactly once per workgroup.
#pragma unroll 1
  for (int mat = 0; mat < 2; ++mat) {
    const u16*   W    = Wbf + (mat ? 2 : 1) * (HIDDEN * HIDDEN);
    const float* bias = mat ? b2 : b1;
    u16*         dst  = mat ? s2 : s1;
#pragma unroll 1
    for (int oi = 0; oi < 2; ++oi) {
      const int oT = oT0 + oi;
      v16bf bw[8];
#pragma unroll
      for (int k = 0; k < 8; ++k) bw[k] = load_b(W, HIDDEN, oT * 16, k * 32, lane);
      const float bv = bias[oT * 16 + ln];
#pragma unroll 1
      for (int r = 0; r < nT; ++r) {
        v8f c = {bv, bv, bv, bv, bv, bv, bv, bv};
#pragma unroll
        for (int k = 0; k < 8; ++k)
          c = wmma_bf16(load_a(xs, PX, r * 16, k * 32, lane), bw[k], c);
#pragma unroll
        for (int v = 0; v < 8; ++v)
          dst[(r * 16 + v + 8 * half) * PX + oT * 16 + ln] = f2bf(c[v]);
      }
    }
  }
  __syncthreads();

  // ---- Phase 2: scores = s1 @ s2^T; one 16-row tile per wave, in registers ----
  v8f F[8];
  if (rT < nT) {
    v16bf a1[8];
#pragma unroll
    for (int k = 0; k < 8; ++k) a1[k] = load_a(s1, PX, rT * 16, k * 32, lane);
#pragma unroll 1
    for (int j = 0; j < nT; ++j) {
      v8f c = {0.f, 0.f, 0.f, 0.f, 0.f, 0.f, 0.f, 0.f};
#pragma unroll
      for (int k = 0; k < 8; ++k)
        c = wmma_bf16(a1[k], load_b(s2, PX, j * 16, k * 32, lane), c);
      F[j] = c;
    }
  }
  __syncthreads();   // all s1/s2 reads done -> regions reusable

  // ---- Phase 3a: scale + diag mask + softmax (lane reductions), write -probs ----
  if (rT < nT) {
#pragma unroll
    for (int v = 0; v < 8; ++v) {
      const int rowIn = v + 8 * half;   // row within tile held by this lane half
      float mx = -3.0e38f;
      for (int j = 0; j < nT; ++j) {
        float s = F[j][v] * 0.0625f;    // SCALE = 1/sqrt(256)
        const int col = j * 16 + ln;
        if (col >= N) s = -3.0e38f;
        else if (j == rT && rowIn == ln) s += -10000.0f;  // eye(n)*NEG
        F[j][v] = s;
        mx = fmaxf(mx, s);
      }
      mx = fmaxf(mx, __shfl_xor(mx, 1));
      mx = fmaxf(mx, __shfl_xor(mx, 2));
      mx = fmaxf(mx, __shfl_xor(mx, 4));
      mx = fmaxf(mx, __shfl_xor(mx, 8));
      float sum = 0.f;
      for (int j = 0; j < nT; ++j) {
        const int col = j * 16 + ln;
        const float e = (col < N) ? __expf(F[j][v] - mx) : 0.f;
        F[j][v] = e;
        sum += e;
      }
      sum += __shfl_xor(sum, 1);
      sum += __shfl_xor(sum, 2);
      sum += __shfl_xor(sum, 4);
      sum += __shfl_xor(sum, 8);
      const float inv = -1.0f / sum;    // store NEGATED probs: out = s0 + (-P)@s3
      for (int j = 0; j < nT; ++j)
        probs[(rT * 16 + rowIn) * PP + j * 16 + ln] = f2bf(F[j][v] * inv);
    }
  }

  // ---- Phase 3b: s3T[o][m] = (x@W3^T + b3) transposed (bf16 over s2 region) ----
  {
    const u16* W3 = Wbf + 3 * (HIDDEN * HIDDEN);
#pragma unroll 1
    for (int oi = 0; oi < 2; ++oi) {
      const int oT = oT0 + oi;
      v16bf bw[8];
#pragma unroll
      for (int k = 0; k < 8; ++k) bw[k] = load_b(W3, HIDDEN, oT * 16, k * 32, lane);
      const float bv = b3[oT * 16 + ln];
#pragma unroll 1
      for (int r = 0; r < nT; ++r) {
        v8f c = {bv, bv, bv, bv, bv, bv, bv, bv};
#pragma unroll
        for (int k = 0; k < 8; ++k)
          c = wmma_bf16(load_a(xs, PX, r * 16, k * 32, lane), bw[k], c);
#pragma unroll
        for (int v = 0; v < 8; ++v)
          s3T[(oT * 16 + ln) * PT + r * 16 + v + 8 * half] = f2bf(c[v]);
      }
    }
  }
  // TERM segment (Npad==16): zero K-pad columns so the K=32 PV WMMA reads zeros
  if (Npad == 16) {
    for (int i = tid; i < 16 * 16; i += 256)
      probs[(i >> 4) * PP + 16 + (i & 15)] = 0;
    for (int i = tid; i < HIDDEN * 16; i += 256)
      s3T[(i >> 4) * PT + 16 + (i & 15)] = 0;
  }
  __syncthreads();

  // ---- Phase 4: out = b0 + x@W0^T + (-P)@s3 (one accumulator chain), f32 store ----
  {
    float* og = out + (b * SEQ + segStart) * HIDDEN;
    const int kmEnd = ((Npad >= 32) ? Npad : 32) >> 5;
#pragma unroll 1
    for (int oi = 0; oi < 2; ++oi) {
      const int oT = oT0 + oi;
      v16bf bw[8];
#pragma unroll
      for (int k = 0; k < 8; ++k) bw[k] = load_b(Wbf, HIDDEN, oT * 16, k * 32, lane);
      v16bf bv3[4];                       // s3T B-fragments: depend on (oT, km) only
      for (int km = 0; km < kmEnd; ++km)
        bv3[km] = load_b(s3T, PT, oT * 16, km * 32, lane);
      const float bv = b0[oT * 16 + ln];
#pragma unroll 1
      for (int r = 0; r < nT; ++r) {
        v8f c = {bv, bv, bv, bv, bv, bv, bv, bv};
#pragma unroll
        for (int k = 0; k < 8; ++k)
          c = wmma_bf16(load_a(xs, PX, r * 16, k * 32, lane), bw[k], c);
        for (int km = 0; km < kmEnd; ++km)
          c = wmma_bf16(load_a(probs, PP, r * 16, km * 32, lane), bv3[km], c);
#pragma unroll
        for (int v = 0; v < 8; ++v) {
          const int row = r * 16 + v + 8 * half;
          if (row < N) og[row * HIDDEN + oT * 16 + ln] = c[v];
        }
      }
    }
  }
}

extern "C" void kernel_launch(void* const* d_in, const int* in_sizes, int n_in,
                              void* d_out, int out_size, void* d_ws, size_t ws_size,
                              hipStream_t stream) {
  const float* toks = (const float*)d_in[0];
  const float* W0 = (const float*)d_in[1]; const float* b0 = (const float*)d_in[2];
  const float* W1 = (const float*)d_in[3]; const float* b1 = (const float*)d_in[4];
  const float* W2 = (const float*)d_in[5]; const float* b2 = (const float*)d_in[6];
  const float* W3 = (const float*)d_in[7]; const float* b3 = (const float*)d_in[8];
  u16* Wbf = (u16*)d_ws;                        // 4*256*256 bf16 = 512 KB

  const int nbatch = in_sizes[0] / (SEQ * HIDDEN);

  wcvt_kernel<<<(4 * HIDDEN * HIDDEN) / 256, 256, 0, stream>>>(W0, W1, W2, W3, Wbf);
  fused_attn_kernel<<<nbatch * NSEG, 256, 0, stream>>>(toks, Wbf, b0, b1, b2, b3,
                                                       (float*)d_out);
}
